// BERT_89017492177566
// MI455X (gfx1250) — matrix operand order repaired and checked by reading, not compiled, not run on hardware
//
#include <hip/hip_runtime.h>
#include <hip/hip_bf16.h>
#include <math.h>

typedef __attribute__((ext_vector_type(16))) _Float16 v16h;
typedef __attribute__((ext_vector_type(8)))  _Float16 v8h;
typedef __attribute__((ext_vector_type(8)))  float    v8f;
typedef __attribute__((ext_vector_type(4)))  unsigned int u32x4;
typedef __attribute__((ext_vector_type(8)))  int      i32x8;
typedef __attribute__((ext_vector_type(4)))  int      i32x4;

#define NEG_BIG (-3.402823466e38f)

#if __has_builtin(__builtin_amdgcn_tensor_load_to_lds)
#define USE_TDM 1
#else
#define USE_TDM 0
#endif

// ---------------------------------------------------------------------------
// Block reductions (256 threads)
// ---------------------------------------------------------------------------
__device__ __forceinline__ float blk_sum(float v, float* red) {
  int t = threadIdx.x;
  red[t] = v; __syncthreads();
  for (int st = 128; st > 0; st >>= 1) {
    if (t < st) red[t] += red[t + st];
    __syncthreads();
  }
  float r = red[0]; __syncthreads();
  return r;
}

__device__ __forceinline__ float blk_max(float v, float* red) {
  int t = threadIdx.x;
  red[t] = v; __syncthreads();
  for (int st = 128; st > 0; st >>= 1) {
    if (t < st) red[t] = fmaxf(red[t], red[t + st]);
    __syncthreads();
  }
  float r = red[0]; __syncthreads();
  return r;
}

__device__ __forceinline__ v16h frag16(const _Float16* p, int kb) {
  v8h lo = *(const v8h*)(p + kb);
  v8h hi = *(const v8h*)(p + kb + 8);
  return __builtin_shufflevector(lo, hi, 0, 1, 2, 3, 4, 5, 6, 7,
                                 8, 9, 10, 11, 12, 13, 14, 15);
}

#if USE_TDM
// ---------------------------------------------------------------------------
// TDM: DMA a 2D tile (rows x 32 halves, row stride = strideElems halves) from
// global memory into LDS at ldsAddr.  D# packing per CDNA5 ISA 8.3/8.4:
// group0: count=1 | lds_addr | global_addr(57b) | type=2
// group1: data_size=2B | tensor_dim0=strideElems | tensor_dim1=rows |
//         tile_dim0=32 | tile_dim1=rows | tensor_dim0_stride=strideElems
// ---------------------------------------------------------------------------
__device__ __forceinline__ void tdm_load_tile(
    const _Float16* gsrc, unsigned ldsAddr, int rows, int strideElems)
{
  unsigned long long ga = (unsigned long long)(size_t)gsrc;
  u32x4 g0;
  g0[0] = 1u;                                         // count=1, user D#
  g0[1] = ldsAddr;                                    // LDS byte address
  g0[2] = (unsigned)(ga & 0xFFFFFFFFu);               // global addr [31:0]
  g0[3] = (unsigned)((ga >> 32) & 0x01FFFFFFu)        // global addr [56:32]
        | (2u << 30);                                 // type = 2 ("image")
  unsigned K = (unsigned)strideElems;
  unsigned R = (unsigned)rows;
  i32x8 g1;
  g1[0] = (int)(1u << 16);                            // data_size = 2 bytes
  g1[1] = (int)((K & 0xFFFFu) << 16);                 // tensor_dim0 [15:0]
  g1[2] = (int)((K >> 16) | ((R & 0xFFFFu) << 16));   // dim0 hi | tensor_dim1 lo
  g1[3] = (int)((R >> 16) | (32u << 16));             // dim1 hi | tile_dim0=32
  g1[4] = (int)(R & 0xFFFFu);                         // tile_dim1 = rows
  g1[5] = (int)K;                                     // tensor_dim0_stride lo32
  g1[6] = 0;                                          // stride hi | dim1_stride
  g1[7] = 0;
  i32x4 z4 = {0, 0, 0, 0};
#if defined(__clang_major__) && (__clang_major__ >= 23)
  i32x8 z8 = {0, 0, 0, 0, 0, 0, 0, 0};
  __builtin_amdgcn_tensor_load_to_lds(g0, g1, z4, z4, z8, 0);   // 6-arg form
#else
  __builtin_amdgcn_tensor_load_to_lds(g0, g1, z4, z4, 0);       // 5-arg form
#endif
}

__device__ __forceinline__ unsigned lds_addr_of(const void* p) {
  return (unsigned)(size_t)p;    // LDS aperture: byte offset in addr[31:0]
}
#endif

// ---------------------------------------------------------------------------
// f32 -> f16 bulk convert (weights)
// ---------------------------------------------------------------------------
__global__ __launch_bounds__(256) void f32tof16_kernel(
    const float* __restrict__ in, _Float16* __restrict__ out, int n)
{
  for (size_t i = (size_t)blockIdx.x * 256 + threadIdx.x; i < (size_t)n;
       i += (size_t)gridDim.x * 256)
    out[i] = (_Float16)in[i];
}

// ---------------------------------------------------------------------------
// Embedding + sinusoidal positional encoding.  One block per (b,s) row.
// ---------------------------------------------------------------------------
__global__ __launch_bounds__(256) void embed_pe_kernel(
    const int* __restrict__ tokens, const float* __restrict__ emb,
    float* __restrict__ x)
{
  const int D = 768;
  int row = blockIdx.x;           // b*S + s
  int s = row & 1023;
  int tok = tokens[row];
  const float* er = emb + (size_t)tok * D;
  float* xr = x + (size_t)row * D;
  for (int d = threadIdx.x; d < D; d += 256) {
    int k = d & ~1;
    float div = __expf((float)k * (-9.210340371976184f / (float)D));
    float arg = (float)s * div;
    float pe = (d & 1) ? __cosf(arg) : __sinf(arg);
    xr[d] = er[d] + pe;
  }
}

// ---------------------------------------------------------------------------
// LayerNorm over last dim (768); writes f16 for the following GEMM.
// ---------------------------------------------------------------------------
__global__ __launch_bounds__(256) void ln_kernel(
    const float* __restrict__ in, const float* __restrict__ sc,
    const float* __restrict__ bi, _Float16* __restrict__ out)
{
  const int D = 768;
  __shared__ float red[256];
  int r = blockIdx.x;
  const float* x = in + (size_t)r * D;
  float s = 0.f;
  for (int d = threadIdx.x; d < D; d += 256) s += x[d];
  float mean = blk_sum(s, red) * (1.0f / D);
  float v = 0.f;
  for (int d = threadIdx.x; d < D; d += 256) { float t = x[d] - mean; v += t * t; }
  float var = blk_sum(v, red) * (1.0f / D);
  float inv = rsqrtf(var + 1e-5f);
  _Float16* o = out + (size_t)r * D;
  for (int d = threadIdx.x; d < D; d += 256)
    o[d] = (_Float16)((x[d] - mean) * inv * sc[d] + bi[d]);
}

// ---------------------------------------------------------------------------
// LDS-blocked WMMA GEMM:  out = act(A[M,K](f16) @ W[N,K]^T(f16) + bias) (+res)
// Block tile 128x64, 8 waves in 4x2, each wave: 2x2 grid of 16x16 WMMA tiles.
// Panels staged via Tensor Data Mover (double buffered, TDM(k+1) overlaps
// WMMA(k)); TENSORcnt waited by wave 0, published via workgroup barrier.
// act: 0=none, 1=exact gelu.  Writes f16 (outH) or f32 (outF).
// ---------------------------------------------------------------------------
__global__ __launch_bounds__(256) void gemm_wmma_kernel(
    const _Float16* __restrict__ A, const _Float16* __restrict__ W,
    const float* __restrict__ bias, const float* __restrict__ res,
    float* __restrict__ outF, _Float16* __restrict__ outH,
    int M, int N, int K, int act)
{
  __shared__ __align__(16) _Float16 As[2][128 * 32];
  __shared__ __align__(16) _Float16 Bs[2][64 * 32];

  int tid = threadIdx.x;
  int lane = tid & 31;
  int wave = tid >> 5;
  int nbn = N >> 6;                      // N / 64
  int m0 = (blockIdx.x / nbn) << 7;      // * 128
  int n0 = (blockIdx.x % nbn) << 6;      // * 64
  int wm = wave >> 1;                    // 0..3
  int wn = wave & 1;                     // 0..1

  int am  = lane & 15;
  int akb = (lane < 16) ? 0 : 16;
  int bnn = lane & 15;
  int bkb = (lane < 16) ? 0 : 16;

  v8f acc[2][2] = {};

#if USE_TDM
  if (wave == 0) {
    tdm_load_tile(A + (size_t)m0 * K, lds_addr_of(As[0]), 128, K);
    tdm_load_tile(W + (size_t)n0 * K, lds_addr_of(Bs[0]), 64, K);
  }
  int buf = 0;
  for (int k0 = 0; k0 < K; k0 += 32) {
    if (wave == 0) __builtin_amdgcn_s_wait_tensorcnt(0);
    __syncthreads();                       // panel `buf` visible to all waves
    if (wave == 0 && (k0 + 32 < K)) {      // prefetch next panel via TDM
      tdm_load_tile(A + (size_t)m0 * K + k0 + 32, lds_addr_of(As[buf ^ 1]), 128, K);
      tdm_load_tile(W + (size_t)n0 * K + k0 + 32, lds_addr_of(Bs[buf ^ 1]), 64, K);
    }
    v16h afr[2], bfr[2];
#pragma unroll
    for (int mi = 0; mi < 2; ++mi)
      afr[mi] = frag16(As[buf] + (wm * 32 + mi * 16 + am) * 32, akb);
#pragma unroll
    for (int ni = 0; ni < 2; ++ni)
      bfr[ni] = *(const v16h*)(Bs[buf] + (wn * 32 + ni * 16 + bnn) * 32 + bkb);
#pragma unroll
    for (int mi = 0; mi < 2; ++mi)
#pragma unroll
      for (int ni = 0; ni < 2; ++ni)
        acc[mi][ni] = __builtin_amdgcn_wmma_f32_16x16x32_f16(
            false, afr[mi], false, bfr[ni], (short)0, acc[mi][ni], false, false);
    __syncthreads();                       // reads of `buf` done before reuse
    buf ^= 1;
  }
#else
  for (int k0 = 0; k0 < K; k0 += 32) {
#pragma unroll
    for (int i = 0; i < 2; ++i) {
      int hidx = (tid * 2 + i) * 8;
      int r = hidx >> 5, c = hidx & 31;
      const _Float16* src = A + (size_t)(m0 + r) * K + k0 + c;
      *(float4*)(As[0] + hidx) = *(const float4*)src;
      if (k0 + 32 < K) __builtin_prefetch(src + 32, 0, 3);
    }
    {
      int hidx = tid * 8;
      int r = hidx >> 5, c = hidx & 31;
      const _Float16* src = W + (size_t)(n0 + r) * K + k0 + c;
      *(float4*)(Bs[0] + hidx) = *(const float4*)src;
    }
    __syncthreads();
    v16h afr[2], bfr[2];
#pragma unroll
    for (int mi = 0; mi < 2; ++mi)
      afr[mi] = frag16(As[0] + (wm * 32 + mi * 16 + am) * 32, akb);
#pragma unroll
    for (int ni = 0; ni < 2; ++ni)
      bfr[ni] = *(const v16h*)(Bs[0] + (wn * 32 + ni * 16 + bnn) * 32 + bkb);
#pragma unroll
    for (int mi = 0; mi < 2; ++mi)
#pragma unroll
      for (int ni = 0; ni < 2; ++ni)
        acc[mi][ni] = __builtin_amdgcn_wmma_f32_16x16x32_f16(
            false, afr[mi], false, bfr[ni], (short)0, acc[mi][ni], false, false);
    __syncthreads();
  }
#endif

#pragma unroll
  for (int mi = 0; mi < 2; ++mi) {
#pragma unroll
    for (int ni = 0; ni < 2; ++ni) {
      int n = n0 + wn * 32 + ni * 16 + (lane & 15);
      int rbase = m0 + wm * 32 + mi * 16 + ((lane < 16) ? 0 : 8);
      float bv = bias ? bias[n] : 0.f;
#pragma unroll
      for (int r = 0; r < 8; ++r) {
        float v = acc[mi][ni][r] + bv;
        if (act == 1) v = 0.5f * v * (1.0f + erff(v * 0.7071067811865476f));
        if (res) v += res[(size_t)(rbase + r) * N + n];
        if (outH) outH[(size_t)(rbase + r) * N + n] = (_Float16)v;
        else      outF[(size_t)(rbase + r) * N + n] = v;
      }
    }
  }
}

// ---------------------------------------------------------------------------
// Band attention for one layer.  Block = (b, 16-query tile), 8 waves.
// qkv (f16) layout: [B,S,3*D]; head h: q at h*64, k at D+h*64, v at 2D+h*64.
// ctx (f16) -> [B,S,D]; allw (f32) -> this layer's [B,S,S] band (rest zeroed).
// Dynamic LDS: sc[16*288] f32 | wsum[16*288] f32 | Vt[64*288] f16  (73728 B)
// ---------------------------------------------------------------------------
__global__ __launch_bounds__(256) void attn_kernel(
    const _Float16* __restrict__ qkv, const int* __restrict__ mask,
    _Float16* __restrict__ ctx, float* __restrict__ allw)
{
  const int S = 1024, D = 768, HD = 64, NH = 12, BL = 128;
  const int RS = 3 * D;
  const int SCW = 288;

  extern __shared__ char smem[];
  float* sc     = (float*)smem;                  // [16][288]
  float* wsum   = sc + 16 * SCW;                 // [16][288]
  _Float16* Vt  = (_Float16*)(wsum + 16 * SCW);  // [64][288] transposed V

  int b  = blockIdx.x >> 6;
  int qt = blockIdx.x & 63;
  int q0 = qt << 4;
  int kmin = (q0 - BL) > 0 ? (q0 - BL) : 0;
  int kmax = (q0 + 16 + BL) < S ? (q0 + 16 + BL) : S;   // exclusive
  int nk   = kmax - kmin;                               // multiple of 16
  int nkt  = nk >> 4;
  int kpad = (nk + 31) & ~31;

  int tid  = threadIdx.x;
  int lane = tid & 31;
  int wave = tid >> 5;

  for (int i = tid; i < 16 * SCW; i += 256) { sc[i] = 0.f; wsum[i] = 0.f; }
  __syncthreads();

  const _Float16* qb = qkv + (size_t)b * S * RS;

  int am  = lane & 15;
  int akb = (lane < 16) ? 0 : 16;
  int bn  = lane & 15;
  int bkb = (lane < 16) ? 0 : 16;

  for (int h = 0; h < NH; ++h) {
    // stage V transposed (zero-padded to 288 keys): Vt[c][key] = V[key][c]
    for (int i = tid; i < HD * SCW; i += 256) {
      int r = i >> 6;        // key 0..287
      int c = i & 63;        // hd col
      _Float16 v = (_Float16)0.f;
      if (r < nk) v = qb[(size_t)(kmin + r) * RS + 2 * D + h * HD + c];
      Vt[c * SCW + r] = v;
    }
    __syncthreads();

    // scores = (Q @ K^T) * 1/8 ; each wave strides key tiles
    for (int kt = wave; kt < nkt; kt += 8) {
      v8f acc = {};
      int key = kmin + kt * 16 + bn;
#pragma unroll
      for (int k0 = 0; k0 < HD; k0 += 32) {
        v16h af = frag16(qb + (size_t)(q0 + am) * RS + h * HD + k0, akb);
        v16h bf = *(const v16h*)(qb + (size_t)key * RS + D + h * HD + k0 + bkb);
        acc = __builtin_amdgcn_wmma_f32_16x16x32_f16(
            false, af, false, bf, (short)0, acc, false, false);
      }
      int n = kt * 16 + (lane & 15);
      bool kok = (mask[b * S + key] != 0);
      int rb = (lane < 16) ? 0 : 8;
#pragma unroll
      for (int r = 0; r < 8; ++r) {
        int q = q0 + rb + r;
        int dlt = q - key; if (dlt < 0) dlt = -dlt;
        bool ok = kok && (dlt <= BL);
        sc[(rb + r) * SCW + n] = ok ? acc[r] * 0.125f : NEG_BIG;
      }
    }
    __syncthreads();

    // per-row softmax over nk keys + accumulate head-average
    if (tid < 16) {
      float* row = sc + tid * SCW;
      float mx = NEG_BIG;
      for (int j = 0; j < nk; ++j) mx = fmaxf(mx, row[j]);
      float s = 0.f;
      for (int j = 0; j < nk; ++j) { float e = __expf(row[j] - mx); row[j] = e; s += e; }
      float inv = 1.0f / s;
      float* wr = wsum + tid * SCW;
      for (int j = 0; j < nk; ++j) {
        row[j] *= inv;
        wr[j] += row[j] * (1.0f / 12.0f);
      }
    }
    __syncthreads();

    // ctx = w @ V ; waves 0..3 own the 4 column tiles of hd=64
    if (wave < 4) {
      int n0 = wave << 4;
      v8f acc = {};
      for (int k0 = 0; k0 < kpad; k0 += 32) {
        v16h af, bf;
        const float* wrow = sc + am * SCW + k0;
#pragma unroll
        for (int i = 0; i < 8; ++i) {
          af[i]     = (_Float16)wrow[akb + i];
          af[i + 8] = (_Float16)wrow[akb + 8 + i];
        }
        bf = *(const v16h*)(Vt + (n0 + bn) * SCW + k0 + bkb);
        acc = __builtin_amdgcn_wmma_f32_16x16x32_f16(
            false, af, false, bf, (short)0, acc, false, false);
      }
      int n = n0 + (lane & 15);
      int rb = (lane < 16) ? 0 : 8;
#pragma unroll
      for (int r = 0; r < 8; ++r)
        ctx[((size_t)b * S + q0 + rb + r) * D + h * HD + n] = (_Float16)acc[r];
    }
    __syncthreads();
  }

  // head-averaged attention weights into the band of allw[b, q, k]
  for (int i = tid; i < 16 * SCW; i += 256) {
    int m = i / SCW, j = i % SCW;
    if (j < nk)
      allw[((size_t)b * S + q0 + m) * S + kmin + j] = wsum[i];
  }
}

// ---------------------------------------------------------------------------
// Attention pooling + tanh.  One block per batch element.
// ---------------------------------------------------------------------------
__global__ __launch_bounds__(256) void pool_kernel(
    const float* __restrict__ x, const int* __restrict__ mask,
    const float* __restrict__ pool_w, const float* __restrict__ pool_b,
    float* __restrict__ pooled)
{
  const int S = 1024, D = 768;
  __shared__ float sc[1024];
  __shared__ float red[256];
  int b = blockIdx.x;
  const float* xb = x + (size_t)b * S * D;
  for (int s = threadIdx.x; s < S; s += 256) {
    float acc = pool_b[0];
    const float* xr = xb + (size_t)s * D;
    for (int d = 0; d < D; ++d) acc += xr[d] * pool_w[d];
    sc[s] = (mask[b * S + s] == 0) ? NEG_BIG : acc;
  }
  __syncthreads();
  float mx = NEG_BIG;
  for (int s = threadIdx.x; s < S; s += 256) mx = fmaxf(mx, sc[s]);
  mx = blk_max(mx, red);
  float sm = 0.f;
  for (int s = threadIdx.x; s < S; s += 256) { float e = __expf(sc[s] - mx); sc[s] = e; sm += e; }
  sm = blk_sum(sm, red);
  float inv = 1.0f / sm;
  __syncthreads();
  for (int d = threadIdx.x; d < D; d += 256) {
    float acc = 0.f;
    for (int s = 0; s < S; ++s) acc += sc[s] * xb[(size_t)s * D + d];
    pooled[b * D + d] = tanhf(acc * inv);
  }
}

// ---------------------------------------------------------------------------
// Classifier head: LN -> tanh(Linear) -> Linear.  One block per batch element.
// ---------------------------------------------------------------------------
__global__ __launch_bounds__(256) void cls_kernel(
    const float* __restrict__ pooled,
    const float* __restrict__ ln_s, const float* __restrict__ ln_b,
    const float* __restrict__ w1, const float* __restrict__ b1,
    const float* __restrict__ w2, const float* __restrict__ b2,
    float* __restrict__ out)
{
  const int D = 768;
  __shared__ float c[768];
  __shared__ float c1[768];
  __shared__ float red[256];
  int b = blockIdx.x;
  const float* p = pooled + (size_t)b * D;
  float s = 0.f;
  for (int d = threadIdx.x; d < D; d += 256) s += p[d];
  float mean = blk_sum(s, red) * (1.0f / D);
  float v = 0.f;
  for (int d = threadIdx.x; d < D; d += 256) { float t = p[d] - mean; v += t * t; }
  float var = blk_sum(v, red) * (1.0f / D);
  float inv = rsqrtf(var + 1e-5f);
  for (int d = threadIdx.x; d < D; d += 256)
    c[d] = (p[d] - mean) * inv * ln_s[d] + ln_b[d];
  __syncthreads();
  for (int n = threadIdx.x; n < D; n += 256) {
    float a = b1[n];
    const float* wr = w1 + (size_t)n * D;
    for (int k = 0; k < D; ++k) a += c[k] * wr[k];
    c1[n] = tanhf(a);
  }
  __syncthreads();
  if (threadIdx.x < 2) {
    float a = b2[threadIdx.x];
    const float* wr = w2 + (size_t)threadIdx.x * D;
    for (int k = 0; k < D; ++k) a += c1[k] * wr[k];
    out[b * 2 + threadIdx.x] = a;
  }
}

// ---------------------------------------------------------------------------
// Host launch
// ---------------------------------------------------------------------------
extern "C" void kernel_launch(void* const* d_in, const int* in_sizes, int n_in,
                              void* d_out, int out_size, void* d_ws, size_t ws_size,
                              hipStream_t stream) {
  (void)in_sizes; (void)n_in; (void)ws_size;
  const int B = 8, S = 1024, D = 768, F = 3072, L = 4, C = 2;
  const size_t BS = (size_t)B * S;

  const int*   tokens = (const int*)d_in[0];
  const int*   mask   = (const int*)d_in[1];
  const float* emb    = (const float*)d_in[2];
  const float* in_w   = (const float*)d_in[3];
  const float* in_b   = (const float*)d_in[4];
  const float* out_w  = (const float*)d_in[5];
  const float* out_b  = (const float*)d_in[6];
  const float* ln1_s  = (const float*)d_in[7];
  const float* ln1_b  = (const float*)d_in[8];
  const float* ln2_s  = (const float*)d_in[9];
  const float* ln2_b  = (const float*)d_in[10];
  const float* ff_w1  = (const float*)d_in[11];
  const float* ff_b1  = (const float*)d_in[12];
  const float* ff_w2  = (const float*)d_in[13];
  const float* ff_b2  = (const float*)d_in[14];
  const float* pool_w = (const float*)d_in[15];
  const float* pool_b = (const float*)d_in[16];
  const float* cls_ln_s = (const float*)d_in[17];
  const float* cls_ln_b = (const float*)d_in[18];
  const float* cls_w1 = (const float*)d_in[19];
  const float* cls_b1 = (const float*)d_in[20];
  const float* cls_w2 = (const float*)d_in[21];
  const float* cls_b2 = (const float*)d_in[22];

  // workspace carving (256B aligned)
  char* cur = (char*)d_ws;
  auto alloc = [&](size_t bytes) -> char* {
    char* p = cur;
    cur += (bytes + 255) & ~(size_t)255;
    return p;
  };
  float*     X      = (float*)alloc(BS * D * 4);          // residual stream
  _Float16*  Hb     = (_Float16*)alloc(BS * D * 2);       // LN output
  _Float16*  QKV    = (_Float16*)alloc(BS * 3 * D * 2);
  _Float16*  ATT    = (_Float16*)alloc(BS * D * 2);
  _Float16*  FFH    = (_Float16*)alloc(BS * F * 2);
  float*     POOLED = (float*)alloc((size_t)B * D * 4);
  _Float16*  Win16  = (_Float16*)alloc((size_t)L * 3 * D * D * 2);
  _Float16*  Wout16 = (_Float16*)alloc((size_t)L * D * D * 2);
  _Float16*  Wff1   = (_Float16*)alloc((size_t)L * F * D * 2);
  _Float16*  Wff2   = (_Float16*)alloc((size_t)L * D * F * 2);

  float* outp = (float*)d_out;
  float* allw = outp + (size_t)B * C;      // [L, B, S, S]

  // zero all_w (band writes fill the rest) and logits region
  (void)hipMemsetAsync(d_out, 0, (size_t)out_size * sizeof(float), stream);

  // weights -> f16 (once per launch)
  f32tof16_kernel<<<2048, 256, 0, stream>>>(in_w,  Win16,  L * 3 * D * D);
  f32tof16_kernel<<<2048, 256, 0, stream>>>(out_w, Wout16, L * D * D);
  f32tof16_kernel<<<2048, 256, 0, stream>>>(ff_w1, Wff1,   L * F * D);
  f32tof16_kernel<<<2048, 256, 0, stream>>>(ff_w2, Wff2,   L * D * F);

  embed_pe_kernel<<<(int)BS, 256, 0, stream>>>(tokens, emb, X);

  const int ATTN_LDS = (16 * 288 * 4) * 2 + 64 * 288 * 2;  // 73728 B

  for (int l = 0; l < L; ++l) {
    ln_kernel<<<(int)BS, 256, 0, stream>>>(X, ln1_s + l * D, ln1_b + l * D, Hb);

    gemm_wmma_kernel<<<64 * (3 * D / 64), 256, 0, stream>>>(
        Hb, Win16 + (size_t)l * 3 * D * D, in_b + (size_t)l * 3 * D,
        nullptr, nullptr, QKV, (int)BS, 3 * D, D, 0);

    attn_kernel<<<B * (S / 16), 256, ATTN_LDS, stream>>>(
        QKV, mask, ATT, allw + (size_t)l * B * S * S);

    gemm_wmma_kernel<<<64 * (D / 64), 256, 0, stream>>>(
        ATT, Wout16 + (size_t)l * D * D, out_b + (size_t)l * D,
        X, X, nullptr, (int)BS, D, D, 0);

    ln_kernel<<<(int)BS, 256, 0, stream>>>(X, ln2_s + l * D, ln2_b + l * D, Hb);

    gemm_wmma_kernel<<<64 * (F / 64), 256, 0, stream>>>(
        Hb, Wff1 + (size_t)l * F * D, ff_b1 + (size_t)l * F,
        nullptr, nullptr, FFH, (int)BS, F, D, 1);

    gemm_wmma_kernel<<<64 * (D / 64), 256, 0, stream>>>(
        FFH, Wff2 + (size_t)l * D * F, ff_b2 + (size_t)l * D,
        X, X, nullptr, (int)BS, D, F, 0);
  }

  pool_kernel<<<B, 256, 0, stream>>>(X, mask, pool_w, pool_b, POOLED);
  cls_kernel<<<B, 256, 0, stream>>>(POOLED, cls_ln_s, cls_ln_b,
                                    cls_w1, cls_b1, cls_w2, cls_b2, outp);
}
